// GCNv2_72112500900241
// MI455X (gfx1250) — matrix operand order repaired and checked by reading, not compiled, not run on hardware
//
#include <hip/hip_runtime.h>

typedef _Float16 v16h __attribute__((ext_vector_type(16)));
typedef _Float16 v8h  __attribute__((ext_vector_type(8)));
typedef float    v8f  __attribute__((ext_vector_type(8)));

// ---------------------------------------------------------------------------
// Degree / symmetric-norm computation:  deg = 1 (self loop) + count(dst);
// dis = rsqrt(deg); norm(e) = dis[src]*dis[dst].
// ---------------------------------------------------------------------------
__global__ void k_deg_init(float* deg, int n) {
  int i = blockIdx.x * blockDim.x + threadIdx.x;
  if (i < n) deg[i] = 1.0f;  // self-loop contribution
}

__global__ void k_deg_count(const int* __restrict__ dst, float* deg, int e) {
  int i = blockIdx.x * blockDim.x + threadIdx.x;
  if (i < e) unsafeAtomicAdd(&deg[dst[i]], 1.0f);
}

__global__ void k_deg_rsqrt(float* deg, int n) {
  int i = blockIdx.x * blockDim.x + threadIdx.x;
  if (i < n) {
    float d = deg[i];
    deg[i] = d > 0.0f ? rsqrtf(d) : 0.0f;
  }
}

// ---------------------------------------------------------------------------
// f32 -> f16 convert (activations feeding WMMA)
// ---------------------------------------------------------------------------
__global__ void k_cvt_f16(const float* __restrict__ x, _Float16* __restrict__ y, int n) {
  int i = blockIdx.x * blockDim.x + threadIdx.x;
  if (i < n) y[i] = (_Float16)x[i];
}

// ---------------------------------------------------------------------------
// Pack W [64 x nOut] (f32, row-major, fan_in x fan_out) into the per-lane
// B-fragment layout for v_wmma_f32_16x16x32_f16.
// B 32x16 16-bit layout: lane l holds column n = l&15; same K order as the
// A fragment within one 8-VGPR fragment ->
// element j of lane l is K = kc*32 + (j>>3)*16 + (l>>4)*8 + (j&7).
// Wp element index = (((ct*2 + kc)*32 + lane)*16 + j).
// ---------------------------------------------------------------------------
__global__ void k_pack_w(const float* __restrict__ W, _Float16* __restrict__ Wp,
                         int nOut, int nColTiles) {
  int tid = blockIdx.x * blockDim.x + threadIdx.x;
  int total = nColTiles << 10;  // nColTiles * 2 * 32 * 16
  if (tid >= total) return;
  int j    = tid & 15;
  int lane = (tid >> 4) & 31;
  int kc   = (tid >> 9) & 1;
  int ct   = tid >> 10;
  int k    = kc * 32 + ((j >> 3) << 4) + ((lane >> 4) << 3) + (j & 7);
  int col  = ct * 16 + (lane & 15);
  float v  = (col < nOut) ? W[k * nOut + col] : 0.0f;
  Wp[tid]  = (_Float16)v;
}

// ---------------------------------------------------------------------------
// WMMA GEMM:  T[nRows x (16*COL_TILES)] = A[nRows x 64] * W
// One wave per 16-row strip: loads A once, then COL_TILES x (2 WMMAs, K=64).
// COL_TILES/LDT are compile-time -> all B-fragment and store offsets become
// instruction-immediate offsets (no 64-bit address chains).
// nRows must be a multiple of 16 (100000 is). Early-exit is wave-uniform so
// EXEC is all-ones at every WMMA (ISA requirement).
// ---------------------------------------------------------------------------
template <int COL_TILES, int LDT>
__global__ void k_gemm_wmma(const _Float16* __restrict__ A,
                            const _Float16* __restrict__ Wp,
                            float* __restrict__ T,
                            int nRows) {
  int w    = threadIdx.x >> 5;
  int lane = threadIdx.x & 31;
  int rt   = blockIdx.x * 8 + w;       // row tile
  int rowTiles = nRows >> 4;
  if (rt >= rowTiles) return;          // uniform per wave

  int r = lane & 15;
  int h = lane >> 4;

  // A fragment (16-bit A 16x32 ISA layout): element j holds
  // K = kc*32 + (j>>3)*16 + h*8 + (j&7)  -> two contiguous 16B loads per frag.
  const _Float16* arow = A + (size_t)(rt * 16 + r) * 64;
  v8h lo0 = *(const v8h*)(arow + h * 8);
  v8h hi0 = *(const v8h*)(arow + 16 + h * 8);
  v8h lo1 = *(const v8h*)(arow + 32 + h * 8);
  v8h hi1 = *(const v8h*)(arow + 48 + h * 8);
  v16h a0, a1;
#pragma unroll
  for (int j = 0; j < 8; ++j) {
    a0[j] = lo0[j]; a0[j + 8] = hi0[j];
    a1[j] = lo1[j]; a1[j + 8] = hi1[j];
  }

  const v16h* wp = (const v16h*)Wp;
  // Per-lane output base; all further offsets are compile-time immediates.
  float* out = T + (size_t)(rt * 16 + h * 8) * LDT + r;

#pragma unroll
  for (int ct = 0; ct < COL_TILES; ++ct) {
    v16h b0 = wp[(ct * 2 + 0) * 32 + lane];
    v16h b1 = wp[(ct * 2 + 1) * 32 + lane];
    v8f c = {};
    c = __builtin_amdgcn_wmma_f32_16x16x32_f16(false, a0, false, b0, (short)0, c, false, false);
    c = __builtin_amdgcn_wmma_f32_16x16x32_f16(false, a1, false, b1, (short)0, c, false, false);
    // C/D layout: lane l col = l&15; VGPR v row = (l>>4)*8 + v.
#pragma unroll
    for (int v = 0; v < 8; ++v) out[v * LDT + ct * 16] = c[v];
  }
}

// ---------------------------------------------------------------------------
// Scatter phase. Self-loops are folded into init: s[i] = t[i] * dis[i]^2.
// ---------------------------------------------------------------------------
__global__ void k_self_init(const float* __restrict__ t, int ldT,
                            float* __restrict__ s, int ldS, int F,
                            const float* __restrict__ dis, int n) {
  int tid = blockIdx.x * blockDim.x + threadIdx.x;
  int i = tid >> 6, f = tid & 63;
  if (i >= n || f >= F) return;
  float d = dis[i];
  s[(size_t)i * ldS + f] = t[(size_t)i * ldT + f] * d * d;
}

// 16 threads per edge, float4 per thread: s[dst] += t[src] * dis[src]*dis[dst]
__global__ void k_scatter(const float* __restrict__ t, int ldT,
                          float* __restrict__ s, int ldS, int F,
                          const int* __restrict__ src, const int* __restrict__ dst,
                          const float* __restrict__ dis, int e) {
  int tid = blockIdx.x * blockDim.x + threadIdx.x;
  int ei = tid >> 4, g = tid & 15;
  if (ei >= e || g * 4 >= F) return;
  int si = src[ei], di = dst[ei];
  float nrm = dis[si] * dis[di];
  const float4 v = *(const float4*)(t + (size_t)si * ldT + g * 4);
  float* o = s + (size_t)di * ldS + g * 4;
  unsafeAtomicAdd(o + 0, v.x * nrm);
  unsafeAtomicAdd(o + 1, v.y * nrm);
  unsafeAtomicAdd(o + 2, v.z * nrm);
  unsafeAtomicAdd(o + 3, v.w * nrm);
}

// h1 = relu(s1 + b1 + res + br) -> f16 (feeds next WMMA GEMM)
__global__ void k_combine1(const float* __restrict__ s, const float* __restrict__ res,
                           const float* __restrict__ b1, const float* __restrict__ br,
                           _Float16* __restrict__ hOut, int n64) {
  int tid = blockIdx.x * blockDim.x + threadIdx.x;
  if (tid >= n64) return;
  int f = tid & 63;
  float v = s[tid] + b1[f] + res[tid] + br[f];
  hOut[tid] = (_Float16)(v > 0.0f ? v : 0.0f);
}

// h2 = relu(s2 + b2) -> f16
__global__ void k_combine2(const float* __restrict__ s, const float* __restrict__ b2,
                           _Float16* __restrict__ hOut, int n64) {
  int tid = blockIdx.x * blockDim.x + threadIdx.x;
  if (tid >= n64) return;
  int f = tid & 63;
  float v = s[tid] + b2[f];
  hOut[tid] = (_Float16)(v > 0.0f ? v : 0.0f);
}

// log_softmax over 40 logits, one wave32 per node (lanes cover cols, +32 tail)
__global__ void k_logsoftmax40(const float* __restrict__ s, const float* __restrict__ b,
                               float* __restrict__ out, int n) {
  int i = blockIdx.x * 8 + (threadIdx.x >> 5);
  int lane = threadIdx.x & 31;
  if (i >= n) return;
  const float* row = s + (size_t)i * 40;
  float v0 = row[lane] + b[lane];               // lane < 32 < 40 always valid
  bool has2 = (lane + 32) < 40;                 // lanes 0..7
  float v1 = has2 ? (row[lane + 32] + b[lane + 32]) : -3.4e38f;

  float m = fmaxf(v0, v1);
#pragma unroll
  for (int off = 16; off; off >>= 1) m = fmaxf(m, __shfl_xor(m, off, 32));
  float sum = __expf(v0 - m) + (has2 ? __expf(v1 - m) : 0.0f);
#pragma unroll
  for (int off = 16; off; off >>= 1) sum += __shfl_xor(sum, off, 32);
  float lse = m + __logf(sum);

  out[(size_t)i * 40 + lane] = v0 - lse;
  if (has2) out[(size_t)i * 40 + lane + 32] = v1 - lse;
}

// ---------------------------------------------------------------------------
extern "C" void kernel_launch(void* const* d_in, const int* in_sizes, int n_in,
                              void* d_out, int out_size, void* d_ws, size_t ws_size,
                              hipStream_t stream) {
  const float* x  = (const float*)d_in[0];
  const float* W1 = (const float*)d_in[1];
  const float* b1 = (const float*)d_in[2];
  const float* W2 = (const float*)d_in[3];
  const float* b2 = (const float*)d_in[4];
  const float* W3 = (const float*)d_in[5];
  const float* b3 = (const float*)d_in[6];
  const float* Wr = (const float*)d_in[7];
  const float* br = (const float*)d_in[8];
  const int*   ei = (const int*)d_in[9];

  const int N = in_sizes[0] / 64;   // 100000
  const int E = in_sizes[9] / 2;    // 1600000
  const int* src = ei;
  const int* dst = ei + E;

  char* wsb = (char*)d_ws;
  size_t off = 0;
  auto alloc = [&](size_t bytes) -> void* {
    void* p = wsb + off;
    off += (bytes + 255) & ~(size_t)255;
    return p;
  };
  float*    dis = (float*)alloc((size_t)N * 4);            // deg -> dis in place
  _Float16* Ah  = (_Float16*)alloc((size_t)N * 64 * 2);    // x / h1 / h2 (f16)
  float*    T   = (float*)alloc((size_t)N * 64 * 4);       // t1 / res / t2 / t3
  float*    S   = (float*)alloc((size_t)N * 64 * 4);       // scatter accumulator
  _Float16* W1p = (_Float16*)alloc(4 * 1024 * 2);
  _Float16* W2p = (_Float16*)alloc(4 * 1024 * 2);
  _Float16* Wrp = (_Float16*)alloc(4 * 1024 * 2);
  _Float16* W3p = (_Float16*)alloc(3 * 1024 * 2);

  const int TB = 256;
  auto g = [&](long total) { return dim3((unsigned)((total + TB - 1) / TB)); };
  const int rowTiles = N / 16;
  const dim3 ggemm((unsigned)((rowTiles + 7) / 8));  // 8 waves/block, 1 row-tile/wave

  // --- normalization ---
  k_deg_init <<<g(N), TB, 0, stream>>>(dis, N);
  k_deg_count<<<g(E), TB, 0, stream>>>(dst, dis, E);
  k_deg_rsqrt<<<g(N), TB, 0, stream>>>(dis, N);

  // --- prep: activations to f16, weights to WMMA fragment layout ---
  k_cvt_f16<<<g((long)N * 64), TB, 0, stream>>>(x, Ah, N * 64);
  k_pack_w <<<g(4 * 1024), TB, 0, stream>>>(W1, W1p, 64, 4);
  k_pack_w <<<g(4 * 1024), TB, 0, stream>>>(W2, W2p, 64, 4);
  k_pack_w <<<g(4 * 1024), TB, 0, stream>>>(Wr, Wrp, 64, 4);
  k_pack_w <<<g(3 * 1024), TB, 0, stream>>>(W3, W3p, 40, 3);

  // --- layer 1: h1 = relu(scatter(x@W1) + b1 + (x@Wr + br)) ---
  k_gemm_wmma<4, 64><<<ggemm, TB, 0, stream>>>(Ah, W1p, T, N);          // t1
  k_self_init<<<g((long)N * 64), TB, 0, stream>>>(T, 64, S, 64, 64, dis, N);
  k_scatter  <<<g((long)E * 16), TB, 0, stream>>>(T, 64, S, 64, 64, src, dst, dis, E);
  k_gemm_wmma<4, 64><<<ggemm, TB, 0, stream>>>(Ah, Wrp, T, N);          // res
  k_combine1 <<<g((long)N * 64), TB, 0, stream>>>(S, T, b1, br, Ah, N * 64);

  // --- layer 2: h2 = relu(scatter(h1@W2) + b2) ---
  k_gemm_wmma<4, 64><<<ggemm, TB, 0, stream>>>(Ah, W2p, T, N);          // t2
  k_self_init<<<g((long)N * 64), TB, 0, stream>>>(T, 64, S, 64, 64, dis, N);
  k_scatter  <<<g((long)E * 16), TB, 0, stream>>>(T, 64, S, 64, 64, src, dst, dis, E);
  k_combine2 <<<g((long)N * 64), TB, 0, stream>>>(S, b2, Ah, N * 64);

  // --- layer 3: logits = scatter(h2@W3) + b3 ; out = log_softmax ---
  k_gemm_wmma<3, 48><<<ggemm, TB, 0, stream>>>(Ah, W3p, T, N);          // t3 (cols 40..47 padded)
  k_self_init<<<g((long)N * 64), TB, 0, stream>>>(T, 48, S, 40, 40, dis, N);
  k_scatter  <<<g((long)E * 16), TB, 0, stream>>>(T, 48, S, 40, 40, src, dst, dis, E);
  k_logsoftmax40<<<dim3((unsigned)((N + 7) / 8)), TB, 0, stream>>>(S, b3, (float*)d_out, N);
}
